// QuantumNATExtended_65481071403227
// MI455X (gfx1250) — compile-verified, matrix operand order checked by reading, not run on hardware
//
#include <hip/hip_runtime.h>
#include <math.h>

#define NQ       10
#define DIM      1024
#define NLAYERS  4
#define BATCH    16384
#define SPB      16              // states (batch elems) per block = one wave each
#define THREADS  (SPB * 32)
#define PSTRIDE  1032            // 1024 + 8 halves pad (keeps 16B align, spreads banks)

typedef __attribute__((ext_vector_type(16))) _Float16 v16h;
typedef __attribute__((ext_vector_type(8)))  _Float16 v8h;
typedef __attribute__((ext_vector_type(8)))  float    v8f;

__device__ __forceinline__ float shflx(float v, int mask) {
  return __shfl_xor(v, mask, 32);
}

// ---------------- single-qubit gates, register-bit (bitpos <= 4) ----------------
template<int STR>
__device__ __forceinline__ void ry_reg(float* ar, float* ai, float c, float s) {
#pragma unroll
  for (int r = 0; r < 32; ++r) {
    if (!(r & STR)) {
      const int q = r | STR;
      float a0r = ar[r], a0i = ai[r], a1r = ar[q], a1i = ai[q];
      ar[r] = c * a0r - s * a1r;  ai[r] = c * a0i - s * a1i;
      ar[q] = s * a0r + c * a1r;  ai[q] = s * a0i + c * a1i;
    }
  }
}

template<int STR>
__device__ __forceinline__ void rot_reg(float* ar, float* ai,
    float m00r, float m00i, float m01r, float m01i,
    float m10r, float m10i, float m11r, float m11i) {
#pragma unroll
  for (int r = 0; r < 32; ++r) {
    if (!(r & STR)) {
      const int q = r | STR;
      float a0r = ar[r], a0i = ai[r], a1r = ar[q], a1i = ai[q];
      ar[r] = m00r * a0r - m00i * a0i + m01r * a1r - m01i * a1i;
      ai[r] = m00r * a0i + m00i * a0r + m01r * a1i + m01i * a1r;
      ar[q] = m10r * a0r - m10i * a0i + m11r * a1r - m11i * a1i;
      ai[q] = m10r * a0i + m10i * a0r + m11r * a1i + m11i * a1r;
    }
  }
}

// ---------------- single-qubit gates, lane-bit (bitpos >= 5) ----------------
__device__ __forceinline__ void ry_lane(float* ar, float* ai, float c, float s,
                                        int lb, int lane) {
  const int  mask = 1 << lb;
  const bool hi   = (lane & mask) != 0;
  const float cp  = hi ? s : -s;   // coeff of partner amplitude
#pragma unroll
  for (int r = 0; r < 32; ++r) {
    float pr = shflx(ar[r], mask), pi = shflx(ai[r], mask);
    ar[r] = c * ar[r] + cp * pr;
    ai[r] = c * ai[r] + cp * pi;
  }
}

__device__ __forceinline__ void rot_lane(float* ar, float* ai, int lb, int lane,
    float m00r, float m00i, float m01r, float m01i,
    float m10r, float m10i, float m11r, float m11i) {
  const int  mask = 1 << lb;
  const bool hi   = (lane & mask) != 0;
  const float cor = hi ? m11r : m00r, coi = hi ? m11i : m00i; // own coeff
  const float cpr = hi ? m10r : m01r, cpi = hi ? m10i : m01i; // partner coeff
#pragma unroll
  for (int r = 0; r < 32; ++r) {
    float pr = shflx(ar[r], mask), pi = shflx(ai[r], mask);
    float vr = ar[r], vi = ai[r];
    ar[r] = cor * vr - coi * vi + cpr * pr - cpi * pi;
    ai[r] = cor * vi + coi * vr + cpr * pi + cpi * pr;
  }
}

// ---------------- CNOTs ----------------
// both bits in-register: pure compile-time register permutation
template<int BCM, int BTM>
__device__ __forceinline__ void cnot_rr(float* ar, float* ai) {
#pragma unroll
  for (int r = 0; r < 32; ++r) {
    if ((r & BCM) && !(r & BTM)) {
      const int q = r | BTM;
      float tr = ar[r], ti = ai[r];
      ar[r] = ar[q]; ai[r] = ai[q];
      ar[q] = tr;    ai[q] = ti;
    }
  }
}

// both bits in lane id: conditional lane swap
__device__ __forceinline__ void cnot_ll(float* ar, float* ai, int lbc, int lbt, int lane) {
  const int  mask = 1 << lbt;
  const bool c    = (lane >> lbc) & 1;
#pragma unroll
  for (int r = 0; r < 32; ++r) {
    float pr = shflx(ar[r], mask), pi = shflx(ai[r], mask);
    ar[r] = c ? pr : ar[r];
    ai[r] = c ? pi : ai[r];
  }
}

// ctrl = lane bit 0, tgt = register bit 4 (wire4 -> wire5): in-lane swap gated by lane
__device__ __forceinline__ void cnot_sl(float* ar, float* ai, int lane) {
  const bool c = lane & 1;
#pragma unroll
  for (int r = 0; r < 16; ++r) {
    float t0r = ar[r], t0i = ai[r], t1r = ar[r + 16], t1i = ai[r + 16];
    ar[r]      = c ? t1r : t0r;  ai[r]      = c ? t1i : t0i;
    ar[r + 16] = c ? t0r : t1r;  ai[r + 16] = c ? t0i : t1i;
  }
}

// ---------------- wire dispatch (wire w <-> index bit 9-w) ----------------
template<int W>
__device__ __forceinline__ void apply_ry(float* ar, float* ai, float ang, int lane) {
  float s, c;
  __sincosf(0.5f * ang, &s, &c);
  constexpr int bp = 9 - W;
  if constexpr (bp <= 4) ry_reg<(1 << bp)>(ar, ai, c, s);
  else                   ry_lane(ar, ai, c, s, bp - 5, lane);
}

template<int W>
__device__ __forceinline__ void apply_rot(const float* __restrict__ params, int L,
                                          float* ar, float* ai, int lane) {
  const int base = (L * NQ + W) * 3;
  float phi = params[base], theta = params[base + 1], omega = params[base + 2];
  float st, ct; __sincosf(0.5f * theta, &st, &ct);
  float sa, ca; __sincosf(0.5f * (phi + omega), &sa, &ca);
  float sb, cb; __sincosf(0.5f * (phi - omega), &sb, &cb);
  // reference: m00=e^{-ia}c, m01=-e^{+ib}s, m10=e^{-ib}s, m11=e^{+ia}c
  float m00r =  ca * ct, m00i = -sa * ct;
  float m01r = -cb * st, m01i = -sb * st;
  float m10r =  cb * st, m10i = -sb * st;
  float m11r =  ca * ct, m11i =  sa * ct;
  constexpr int bp = 9 - W;
  if constexpr (bp <= 4)
    rot_reg<(1 << bp)>(ar, ai, m00r, m00i, m01r, m01i, m10r, m10i, m11r, m11i);
  else
    rot_lane(ar, ai, bp - 5, lane, m00r, m00i, m01r, m01i, m10r, m10i, m11r, m11i);
}

template<int I>   // CNOT ctrl=I, tgt=I+1 ; bits bc=9-I, bt=8-I
__device__ __forceinline__ void apply_cnot(float* ar, float* ai, int lane) {
  if constexpr (I <= 3)      cnot_ll(ar, ai, 4 - I, 3 - I, lane);
  else if constexpr (I == 4) cnot_sl(ar, ai, lane);
  else                       cnot_rr<(1 << (9 - I)), (1 << (8 - I))>(ar, ai);
}

// ================= kernel 1: fused circuit + WMMA readout =================
__global__ __launch_bounds__(THREADS)
void qnat_circuit(const float* __restrict__ x, const float* __restrict__ enc_w,
                  const float* __restrict__ enc_b, const float* __restrict__ params,
                  float* __restrict__ zout) {
  const int lane = threadIdx.x & 31;
  const int wave = threadIdx.x >> 5;
  const int b    = blockIdx.x * SPB + wave;

  __shared__ _Float16 P[SPB][PSTRIDE];   // probs, column-per-state, f16, 33 KB

  // ----- feature: mean of x[b,0:64] -----
  float v = x[b * 64 + lane] + x[b * 64 + 32 + lane];
#pragma unroll
  for (int o = 16; o >= 1; o >>= 1) v += shflx(v, o);
  const float scalar = v * (1.0f / 64.0f);

  // ----- state |0>: amp index k = lane*32 + r -----
  float ar[32], ai[32];
#pragma unroll
  for (int r = 0; r < 32; ++r) { ar[r] = 0.f; ai[r] = 0.f; }
  ar[0] = (lane == 0) ? 1.f : 0.f;

  // ----- RY encoding -----
  apply_ry<0>(ar, ai, scalar * enc_w[0] + enc_b[0], lane);
  apply_ry<1>(ar, ai, scalar * enc_w[1] + enc_b[1], lane);
  apply_ry<2>(ar, ai, scalar * enc_w[2] + enc_b[2], lane);
  apply_ry<3>(ar, ai, scalar * enc_w[3] + enc_b[3], lane);
  apply_ry<4>(ar, ai, scalar * enc_w[4] + enc_b[4], lane);
  apply_ry<5>(ar, ai, scalar * enc_w[5] + enc_b[5], lane);
  apply_ry<6>(ar, ai, scalar * enc_w[6] + enc_b[6], lane);
  apply_ry<7>(ar, ai, scalar * enc_w[7] + enc_b[7], lane);
  apply_ry<8>(ar, ai, scalar * enc_w[8] + enc_b[8], lane);
  apply_ry<9>(ar, ai, scalar * enc_w[9] + enc_b[9], lane);

  // ----- variational layers (runtime loop keeps code size sane) -----
  for (int L = 0; L < NLAYERS; ++L) {
    apply_rot<0>(params, L, ar, ai, lane);
    apply_rot<1>(params, L, ar, ai, lane);
    apply_rot<2>(params, L, ar, ai, lane);
    apply_rot<3>(params, L, ar, ai, lane);
    apply_rot<4>(params, L, ar, ai, lane);
    apply_rot<5>(params, L, ar, ai, lane);
    apply_rot<6>(params, L, ar, ai, lane);
    apply_rot<7>(params, L, ar, ai, lane);
    apply_rot<8>(params, L, ar, ai, lane);
    apply_rot<9>(params, L, ar, ai, lane);
    apply_cnot<0>(ar, ai, lane);
    apply_cnot<1>(ar, ai, lane);
    apply_cnot<2>(ar, ai, lane);
    apply_cnot<3>(ar, ai, lane);
    apply_cnot<4>(ar, ai, lane);
    apply_cnot<5>(ar, ai, lane);
    apply_cnot<6>(ar, ai, lane);
    apply_cnot<7>(ar, ai, lane);
    apply_cnot<8>(ar, ai, lane);
  }

  // ----- stage probabilities into LDS: P[state][k], k = lane*32 + r -----
#pragma unroll
  for (int r = 0; r < 32; ++r) {
    float p = ar[r] * ar[r] + ai[r] * ai[r];
    P[wave][lane * 32 + r] = (_Float16)p;
  }
  __syncthreads();

  // ----- readout GEMM: Z(16x16) = S(16x1024,+/-1) * P(1024x16) via 32 WMMAs -----
  if (wave == 0) {
    const int m   = lane & 15;    // row (wire) for A; column (state) for B
    const int grp = lane >> 4;
    // A fragment base: half h <-> K = h + (h&8) + 8*grp (documented 16-bit striping)
    v16h abase;
#pragma unroll
    for (int h = 0; h < 16; ++h) {
      const int kl = h + (h & 8) + (grp << 3);     // k bits [4:0]
      _Float16 val;
      if (m >= NQ)      val = (_Float16)0.0f;
      else if (m >= 5)  val = ((kl >> (9 - m)) & 1) ? (_Float16)-1.0f : (_Float16)1.0f;
      else              val = (_Float16)1.0f;       // sign comes from chunk bits
      abase[h] = val;
    }
    v8f acc = {};
#pragma unroll
    for (int chunk = 0; chunk < 32; ++chunk) {
      _Float16 sgn = (_Float16)1.0f;               // k bits [9:5] == chunk bits [4:0]
      if (m < 5 && ((chunk >> (4 - m)) & 1)) sgn = (_Float16)-1.0f;
      v16h a;
#pragma unroll
      for (int h = 0; h < 16; ++h) a[h] = abase[h] * sgn;

      const int k0 = chunk * 32 + (grp << 3);
      v8h blo = *(const v8h*)(&P[m][k0]);
      v8h bhi = *(const v8h*)(&P[m][k0 + 16]);
      v16h bb;
#pragma unroll
      for (int j = 0; j < 8; ++j) { bb[j] = blo[j]; bb[8 + j] = bhi[j]; }

      acc = __builtin_amdgcn_wmma_f32_16x16x32_f16(false, a, false, bb,
                                                   (short)0, acc, false, false);
    }
    // D layout: lanes 0-15 hold M=0..7 (vgpr j) for state N=lane; lanes 16-31 hold M=8..15
    const int bbase = blockIdx.x * SPB;
    if (grp == 0) {
#pragma unroll
      for (int j = 0; j < 8; ++j) zout[(bbase + m) * NQ + j] = acc[j];
    } else {
      zout[(bbase + m) * NQ + 8] = acc[0];
      zout[(bbase + m) * NQ + 9] = acc[1];
    }
  }
}

// ================= kernel 2: deterministic batch-norm (in place) =================
__global__ __launch_bounds__(256)
void qnat_bn(float* __restrict__ z, const float* __restrict__ gamma,
             const float* __restrict__ beta) {
  const int m   = blockIdx.x;      // wire / column
  const int tid = threadIdx.x;
  __shared__ float sA[256];
  __shared__ float sB[256];

  float s = 0.f, s2 = 0.f;
  for (int b = tid; b < BATCH; b += 256) {
    float v = z[b * NQ + m];
    s += v; s2 += v * v;
  }
  sA[tid] = s; sB[tid] = s2;
  __syncthreads();
  for (int o = 128; o > 0; o >>= 1) {
    if (tid < o) { sA[tid] += sA[tid + o]; sB[tid] += sB[tid + o]; }
    __syncthreads();
  }
  const float mean = sA[0] * (1.0f / BATCH);
  const float var  = sB[0] * (1.0f / BATCH) - mean * mean;
  const float inv  = rsqrtf(var + 1e-5f);
  const float g = gamma[m], bt = beta[m];
  for (int b = tid; b < BATCH; b += 256) {
    float v = z[b * NQ + m];
    z[b * NQ + m] = g * (v - mean) * inv + bt;
  }
}

extern "C" void kernel_launch(void* const* d_in, const int* in_sizes, int n_in,
                              void* d_out, int out_size, void* d_ws, size_t ws_size,
                              hipStream_t stream) {
  const float* x      = (const float*)d_in[0];
  const float* enc_w  = (const float*)d_in[1];
  const float* enc_b  = (const float*)d_in[2];
  const float* params = (const float*)d_in[3];
  const float* gamma  = (const float*)d_in[4];
  const float* beta   = (const float*)d_in[5];
  float* out = (float*)d_out;

  qnat_circuit<<<BATCH / SPB, THREADS, 0, stream>>>(x, enc_w, enc_b, params, out);
  qnat_bn<<<NQ, 256, 0, stream>>>(out, gamma, beta);
}
